// Net_51866025066825
// MI455X (gfx1250) — matrix operand order repaired and checked by reading, not compiled, not run on hardware
//
#include <hip/hip_runtime.h>
#include <hip/hip_bf16.h>
#include <math.h>

// ---------------- problem constants (from reference) ----------------
#define NN    20000     // nodes
#define NE    320000    // edges
#define INDIM 128
#define HIDD  256
#define NEG_SLOPE 0.2f
#define BN_EPS 1e-5f

typedef __attribute__((ext_vector_type(16))) __bf16 v16bf;
typedef __attribute__((ext_vector_type(8)))  __bf16 v8bf;
typedef __attribute__((ext_vector_type(8)))  float  v8f;

__device__ inline v16bf combine16(v8bf lo, v8bf hi) {
    union { v16bf v; v8bf h[2]; } u;
    u.h[0] = lo; u.h[1] = hi;
    return u.v;
}

// ---------------- small helpers ----------------
__global__ void set_val_k(float* p, int n, float v) {
    int i = blockIdx.x * blockDim.x + threadIdx.x;
    if (i < n) p[i] = v;
}

__global__ void add2_k(float* d, const float* a, const float* b, int n) {
    int i = blockIdx.x * blockDim.x + threadIdx.x;
    if (i < n) d[i] = a[i] + b[i];
}

__global__ void relu_k(float* p, long long n) {
    long long i = blockIdx.x * (long long)blockDim.x + threadIdx.x;
    if (i < n) p[i] = fmaxf(p[i], 0.f);
}

// ---------------- batchnorm ----------------
__global__ void bn_reduce_k(const float* __restrict__ x, float* s, float* q, int Nn, int D) {
    int col = blockIdx.x * blockDim.x + threadIdx.x;
    if (col >= D) return;
    int r0 = blockIdx.y * 128;
    int r1 = r0 + 128; if (r1 > Nn) r1 = Nn;
    float ss = 0.f, qq = 0.f;
    for (int r = r0; r < r1; ++r) {
        float v = x[(size_t)r * D + col];
        ss += v; qq += v * v;
    }
    atomicAdd(&s[col], ss);
    atomicAdd(&q[col], qq);
}

__global__ void bn_fin_k(const float* s, const float* q, const float* g, const float* b,
                         float* scale, float* shift, int Nn, int D) {
    int c = blockIdx.x * blockDim.x + threadIdx.x;
    if (c >= D) return;
    float m  = s[c] / (float)Nn;
    float vv = q[c] / (float)Nn - m * m;
    float rs = rsqrtf(vv + BN_EPS);
    float sc = g[c] * rs;
    scale[c] = sc;
    shift[c] = b[c] - m * sc;
}

__global__ void bn_apply_k(const float* __restrict__ x, const float* __restrict__ scale,
                           const float* __restrict__ shift, float* xn, long long total, int D) {
    long long i = blockIdx.x * (long long)blockDim.x + threadIdx.x;
    if (i >= total) return;
    int c = (int)(i % D);
    xn[i] = x[i] * scale[c] + shift[c];
}

// ---------------- WMMA bf16 GEMM: C = A(MxK) @ B(KxN) [+ bias[col]] ----------------
// M%32==0, N%128==0, K%32==0. Block = 256 threads = 8 waves, tile 32x128,
// each wave owns a 16x32 sub-tile (2 wmma / K-step, shared A fragment).
__global__ __launch_bounds__(256)
void gemm_wmma_bf16(const float* __restrict__ A, const float* __restrict__ B,
                    const float* __restrict__ bias, float* __restrict__ C,
                    int M, int K, int N) {
    __shared__ __align__(16) __bf16 As[32][40];   // [row][k]  (padded)
    __shared__ __align__(16) __bf16 Bs[128][40];  // [col][k]  (transposed, padded)
    const int tid  = threadIdx.x;
    const int wave = tid >> 5;
    const int lane = tid & 31;
    const int wm = wave >> 2;          // 0..1  (M sub-tile)
    const int wn = wave & 3;           // 0..3  (32-col group)
    const int row0 = blockIdx.y * 32;
    const int col0 = blockIdx.x * 128;

    v8f acc0 = {}, acc1 = {};

    for (int kb = 0; kb < K; kb += 32) {
        // stage A tile (32 rows x 32 K), 4 floats / thread, coalesced
        {
            int r = tid >> 3;
            int c = (tid & 7) * 4;
            const float4 v = *(const float4*)(A + (size_t)(row0 + r) * K + kb + c);
            As[r][c + 0] = (__bf16)v.x; As[r][c + 1] = (__bf16)v.y;
            As[r][c + 2] = (__bf16)v.z; As[r][c + 3] = (__bf16)v.w;
        }
        // stage B tile (32 K x 128 cols) transposed into Bs[col][k]
        {
            int r = tid >> 3;            // K row 0..31
            int c = (tid & 7) * 16;      // col group
            const float* g = B + (size_t)(kb + r) * N + col0 + c;
            const float4 v0 = *(const float4*)(g + 0);
            const float4 v1 = *(const float4*)(g + 4);
            const float4 v2 = *(const float4*)(g + 8);
            const float4 v3 = *(const float4*)(g + 12);
            Bs[c +  0][r] = (__bf16)v0.x; Bs[c +  1][r] = (__bf16)v0.y;
            Bs[c +  2][r] = (__bf16)v0.z; Bs[c +  3][r] = (__bf16)v0.w;
            Bs[c +  4][r] = (__bf16)v1.x; Bs[c +  5][r] = (__bf16)v1.y;
            Bs[c +  6][r] = (__bf16)v1.z; Bs[c +  7][r] = (__bf16)v1.w;
            Bs[c +  8][r] = (__bf16)v2.x; Bs[c +  9][r] = (__bf16)v2.y;
            Bs[c + 10][r] = (__bf16)v2.z; Bs[c + 11][r] = (__bf16)v2.w;
            Bs[c + 12][r] = (__bf16)v3.x; Bs[c + 13][r] = (__bf16)v3.y;
            Bs[c + 14][r] = (__bf16)v3.z; Bs[c + 15][r] = (__bf16)v3.w;
        }
        __syncthreads();

        // A fragment: ISA 16-bit A 16x32 layout.
        // lanes 0-15: row=lane,   halves 0-7 = K 0..7,  halves 8-15 = K 16..23
        // lanes16-31: row=lane-16, halves 0-7 = K 8..15, halves 8-15 = K 24..31
        const int r  = wm * 16 + (lane & 15);
        const int kl = (lane < 16) ? 0 : 8;
        const v8bf a0 = *(const v8bf*)&As[r][kl];
        const v8bf a1 = *(const v8bf*)&As[r][kl + 16];
        const v16bf af = combine16(a0, a1);

        // B fragments: lane holds column, same K order as A
        // lanes 0-15: K 0..7,16..23 ; lanes 16-31: K 8..15,24..31
        const int kl2 = (lane < 16) ? 0 : 8;
        {
            const int cc = wn * 32 + (lane & 15);
            const v8bf b0 = *(const v8bf*)&Bs[cc][kl2];
            const v8bf b1 = *(const v8bf*)&Bs[cc][kl2 + 16];
            acc0 = __builtin_amdgcn_wmma_f32_16x16x32_bf16(false, af, false, combine16(b0, b1),
                                                           (short)0, acc0, false, false);
        }
        {
            const int cc = wn * 32 + 16 + (lane & 15);
            const v8bf b0 = *(const v8bf*)&Bs[cc][kl2];
            const v8bf b1 = *(const v8bf*)&Bs[cc][kl2 + 16];
            acc1 = __builtin_amdgcn_wmma_f32_16x16x32_bf16(false, af, false, combine16(b0, b1),
                                                           (short)0, acc1, false, false);
        }
        __syncthreads();
    }

    // store: lanes 0-15 -> M rows 0..7, lanes 16-31 -> rows 8..15
    const int rbase = row0 + wm * 16 + ((lane < 16) ? 0 : 8);
    {
        const int col = col0 + wn * 32 + (lane & 15);
        const float bv = bias ? bias[col] : 0.f;
#pragma unroll
        for (int i = 0; i < 8; ++i)
            C[(size_t)(rbase + i) * N + col] = acc0[i] + bv;
    }
    {
        const int col = col0 + wn * 32 + 16 + (lane & 15);
        const float bv = bias ? bias[col] : 0.f;
#pragma unroll
        for (int i = 0; i < 8; ++i)
            C[(size_t)(rbase + i) * N + col] = acc1[i] + bv;
    }
}

// ---------------- tiny scalar GEMM (layer 2, dout=2) ----------------
__global__ void gemm_small_k(const float* __restrict__ A, const float* __restrict__ B,
                             const float* __restrict__ bias, float* C, int M, int K, int N) {
    int gid = blockIdx.x * blockDim.x + threadIdx.x;
    if (gid >= M * N) return;
    int r = gid / N, c = gid % N;
    float s = bias ? bias[c] : 0.f;
    for (int k = 0; k < K; ++k)
        s += A[(size_t)r * K + k] * B[(size_t)k * N + c];
    C[gid] = s;
}

// ---------------- attention scalars ----------------
// one wave per (node, head): a_src/a_dst dot products over C channels
__global__ void node_att_k(const float* __restrict__ hb, const float* __restrict__ att_s,
                           const float* __restrict__ att_d, float* a_s, float* a_d,
                           int Nn, int H, int C) {
    int gw = (blockIdx.x * blockDim.x + threadIdx.x) >> 5;
    int lane = threadIdx.x & 31;
    if (gw >= Nn * H) return;
    int n = gw / H, h = gw % H;
    const float* hp = hb + (size_t)n * H * C + (size_t)h * C;
    float ss = 0.f, sd = 0.f;
    for (int c = lane; c < C; c += 32) {
        float v = hp[c];
        ss += v * att_s[h * C + c];
        sd += v * att_d[h * C + c];
    }
    for (int o = 16; o > 0; o >>= 1) {
        ss += __shfl_xor(ss, o, 32);
        sd += __shfl_xor(sd, o, 32);
    }
    if (lane == 0) { a_s[gw] = ss; a_d[gw] = sd; }
}

// fold W_e (256 x H*C) with att_edge (H x C) -> V (256 x H)
__global__ void fold_V_k(const float* __restrict__ W_e, const float* __restrict__ att_e,
                         float* V, int H, int C) {
    int gid = blockIdx.x * blockDim.x + threadIdx.x;
    if (gid >= HIDD * H) return;
    int k = gid / H, h = gid % H;
    float s = 0.f;
    for (int c = 0; c < C; ++c)
        s += W_e[(size_t)k * H * C + h * C + c] * att_e[h * C + c];
    V[k * H + h] = s;
}

// A (272 x H) = edge_mlp_W (272 x 256) @ V (256 x H)
__global__ void fold_A_k(const float* __restrict__ Wm, const float* __restrict__ V,
                         float* Af, int H) {
    int gid = blockIdx.x * blockDim.x + threadIdx.x;
    if (gid >= 272 * H) return;
    int k = gid / H, h = gid % H;
    float s = 0.f;
    for (int j = 0; j < HIDD; ++j)
        s += Wm[(size_t)k * HIDD + j] * V[j * H + h];
    Af[k * H + h] = s;
}

__global__ void fold_b_k(const float* __restrict__ bm, const float* __restrict__ V,
                         float* bc, int H) {
    int h = blockIdx.x * blockDim.x + threadIdx.x;
    if (h >= H) return;
    float s = 0.f;
    for (int j = 0; j < HIDD; ++j) s += bm[j] * V[j * H + h];
    bc[h] = s;
}

// a_e[e,h] = bconst[h] + edge_feat(16) . A[0:16] + cos(t*tw+tb)(256) . A[16:272]
__global__ void edge_ae_k(const float* __restrict__ eattr, const float* __restrict__ tw,
                          const float* __restrict__ tb, const float* __restrict__ Af,
                          const float* __restrict__ bc, float* ae, int E, int H) {
    int e = blockIdx.x * blockDim.x + threadIdx.x;
    if (e >= E) return;
    float acc[4];
    for (int h = 0; h < H; ++h) acc[h] = bc[h];
    const float* row = eattr + (size_t)e * 17;
    float t = row[0];
    for (int k = 0; k < 16; ++k) {
        float f = row[1 + k];
        const float* ar = Af + k * H;
        for (int h = 0; h < H; ++h) acc[h] += f * ar[h];
    }
    for (int j = 0; j < HIDD; ++j) {
        float cv = cosf(fmaf(t, tw[j], tb[j]));
        const float* ar = Af + (16 + j) * H;
        for (int h = 0; h < H; ++h) acc[h] += cv * ar[h];
    }
    for (int h = 0; h < H; ++h) ae[(size_t)e * H + h] = acc[h];
}

// ---------------- segment softmax ----------------
__device__ inline void atomicMaxF(float* a, float v) {
    if (v >= 0.f) atomicMax((int*)a, __float_as_int(v));
    else          atomicMin((unsigned int*)a, __float_as_uint(v));
}

__global__ void edge_logit_k(const float* __restrict__ as_, const float* __restrict__ ad_,
                             const float* __restrict__ ae, const int* __restrict__ src,
                             const int* __restrict__ dst, float* alp, float* amax,
                             int EH, int H) {
    int gid = blockIdx.x * blockDim.x + threadIdx.x;
    if (gid >= EH) return;
    int e = gid / H, h = gid % H;
    float z = as_[src[e] * H + h] + ad_[dst[e] * H + h] + ae[gid];
    z = z > 0.f ? z : NEG_SLOPE * z;
    alp[gid] = z;
    atomicMaxF(&amax[dst[e] * H + h], z);
}

__global__ void edge_exp_k(float* alp, const float* __restrict__ amax,
                           const int* __restrict__ dst, float* denom, int EH, int H) {
    int gid = blockIdx.x * blockDim.x + threadIdx.x;
    if (gid >= EH) return;
    int e = gid / H, h = gid % H;
    float z = expf(alp[gid] - amax[dst[e] * H + h]);
    alp[gid] = z;
    atomicAdd(&denom[dst[e] * H + h], z);
}

// out[dst, j] += h[src, j] * alpha_norm   (coalesced gather + atomic scatter)
__global__ void scatter_k(const float* __restrict__ hb, const int* __restrict__ src,
                          const int* __restrict__ dst, const float* __restrict__ alp,
                          const float* __restrict__ denom, float* out,
                          long long total, int HC, int C, int H) {
    long long gid = blockIdx.x * (long long)blockDim.x + threadIdx.x;
    if (gid >= total) return;
    int e = (int)(gid / HC);
    int j = (int)(gid - (long long)e * HC);
    int h = j / C;
    int s = src[e], d = dst[e];
    float a = alp[(size_t)e * H + h] / (denom[(size_t)d * H + h] + 1e-16f);
    atomicAdd(&out[(size_t)d * HC + j], hb[(size_t)s * HC + j] * a);
}

// ---------------- host-side layer driver ----------------
static inline int ceil_div(long long a, int b) { return (int)((a + b - 1) / b); }

struct Scratch {
    float *XN, *HB, *OA, *OB;
    float *asrc, *adst, *amax, *denom, *ae, *alph;
    float *Vf, *Af, *bc, *csum, *csq, *scale, *shift, *cbias;
};

static void run_layer(const float* x, int din, int H, int C,
                      void* const* d_in, int pbase,
                      const float* eattr, const float* tw, const float* tb,
                      const float* emW, const float* emb,
                      const int* src, const int* dst,
                      float* out, const Scratch& S, hipStream_t stream) {
    const float* bn_g   = (const float*)d_in[pbase + 0];
    const float* bn_b   = (const float*)d_in[pbase + 1];
    const float* skip_W = (const float*)d_in[pbase + 2];
    const float* skip_b = (const float*)d_in[pbase + 3];
    const float* Wl     = (const float*)d_in[pbase + 4];
    const float* W_e    = (const float*)d_in[pbase + 5];
    const float* att_s  = (const float*)d_in[pbase + 6];
    const float* att_d  = (const float*)d_in[pbase + 7];
    const float* att_e  = (const float*)d_in[pbase + 8];
    const float* bias   = (const float*)d_in[pbase + 9];
    const int dout = H * C;

    // batchnorm
    set_val_k<<<ceil_div(din, 256), 256, 0, stream>>>(S.csum, din, 0.f);
    set_val_k<<<ceil_div(din, 256), 256, 0, stream>>>(S.csq, din, 0.f);
    dim3 gr(ceil_div(din, 128), ceil_div(NN, 128));
    bn_reduce_k<<<gr, 128, 0, stream>>>(x, S.csum, S.csq, NN, din);
    bn_fin_k<<<ceil_div(din, 256), 256, 0, stream>>>(S.csum, S.csq, bn_g, bn_b, S.scale, S.shift, NN, din);
    long long tot_x = (long long)NN * din;
    bn_apply_k<<<ceil_div(tot_x, 256), 256, 0, stream>>>(x, S.scale, S.shift, S.XN, tot_x, din);

    // combined per-column bias: skip_b + layer bias
    add2_k<<<ceil_div(dout, 256), 256, 0, stream>>>(S.cbias, skip_b, bias, dout);

    // h = xn @ W ; out = xn @ skip_W + cbias
    if (dout % 128 == 0) {
        dim3 gg(dout / 128, NN / 32);
        gemm_wmma_bf16<<<gg, 256, 0, stream>>>(S.XN, Wl, nullptr, S.HB, NN, din, dout);
        gemm_wmma_bf16<<<gg, 256, 0, stream>>>(S.XN, skip_W, S.cbias, out, NN, din, dout);
    } else {
        gemm_small_k<<<ceil_div((long long)NN * dout, 256), 256, 0, stream>>>(S.XN, Wl, nullptr, S.HB, NN, din, dout);
        gemm_small_k<<<ceil_div((long long)NN * dout, 256), 256, 0, stream>>>(S.XN, skip_W, S.cbias, out, NN, din, dout);
    }

    // attention scalars
    node_att_k<<<ceil_div((long long)NN * H * 32, 256), 256, 0, stream>>>(S.HB, att_s, att_d, S.asrc, S.adst, NN, H, C);
    fold_V_k<<<ceil_div(HIDD * H, 256), 256, 0, stream>>>(W_e, att_e, S.Vf, H, C);
    fold_A_k<<<ceil_div(272 * H, 256), 256, 0, stream>>>(emW, S.Vf, S.Af, H);
    fold_b_k<<<1, 64, 0, stream>>>(emb, S.Vf, S.bc, H);
    edge_ae_k<<<ceil_div(NE, 256), 256, 0, stream>>>(eattr, tw, tb, S.Af, S.bc, S.ae, NE, H);

    // segment softmax
    int NH = NN * H, EH = NE * H;
    set_val_k<<<ceil_div(NH, 256), 256, 0, stream>>>(S.amax, NH, -INFINITY);
    set_val_k<<<ceil_div(NH, 256), 256, 0, stream>>>(S.denom, NH, 0.f);
    edge_logit_k<<<ceil_div(EH, 256), 256, 0, stream>>>(S.asrc, S.adst, S.ae, src, dst, S.alph, S.amax, EH, H);
    edge_exp_k<<<ceil_div(EH, 256), 256, 0, stream>>>(S.alph, S.amax, dst, S.denom, EH, H);

    // weighted message scatter + relu
    long long tot_m = (long long)NE * dout;
    scatter_k<<<ceil_div(tot_m, 256), 256, 0, stream>>>(S.HB, src, dst, S.alph, S.denom, out, tot_m, dout, C, H);
    relu_k<<<ceil_div((long long)NN * dout, 256), 256, 0, stream>>>(out, (long long)NN * dout);
}

extern "C" void kernel_launch(void* const* d_in, const int* in_sizes, int n_in,
                              void* d_out, int out_size, void* d_ws, size_t ws_size,
                              hipStream_t stream) {
    const float* x0    = (const float*)d_in[0];
    const int*   ei    = (const int*)d_in[1];
    const float* eattr = (const float*)d_in[2];
    const float* tw    = (const float*)d_in[3];
    const float* tb    = (const float*)d_in[4];
    const float* emW   = (const float*)d_in[5];
    const float* emb   = (const float*)d_in[6];
    const int* src = ei;
    const int* dst = ei + NE;

    // workspace layout (floats)
    float* W = (float*)d_ws;
    const size_t NB = (size_t)NN * 1024;
    Scratch S;
    S.XN = W;            S.HB = W + NB;
    S.OA = W + 2 * NB;   S.OB = W + 3 * NB;
    float* p = W + 4 * NB;
    S.asrc = p;  p += (size_t)NN * 4;
    S.adst = p;  p += (size_t)NN * 4;
    S.amax = p;  p += (size_t)NN * 4;
    S.denom = p; p += (size_t)NN * 4;
    S.ae = p;    p += (size_t)NE * 4;
    S.alph = p;  p += (size_t)NE * 4;
    S.Vf = p;    p += HIDD * 4;
    S.Af = p;    p += 272 * 4;
    S.bc = p;    p += 16;
    S.csum = p;  p += 1024;
    S.csq = p;   p += 1024;
    S.scale = p; p += 1024;
    S.shift = p; p += 1024;
    S.cbias = p; p += 1024;

    // layer 0: x (20000x128) -> OA (20000x1024)
    run_layer(x0, INDIM, 4, 256, d_in, 7, eattr, tw, tb, emW, emb, src, dst, S.OA, S, stream);
    // layer 1: OA -> OB (20000x1024)
    run_layer(S.OA, 1024, 4, 256, d_in, 17, eattr, tw, tb, emW, emb, src, dst, S.OB, S, stream);
    // layer 2: OB -> d_out (20000x2)
    run_layer(S.OB, 1024, 1, 2, d_in, 27, eattr, tw, tb, emW, emb, src, dst, (float*)d_out, S, stream);
}
